// OuterMean_19980187861614
// MI455X (gfx1250) — 16-bit and 8-bit matrix operand order repaired and checked by reading, compiled, not run on hardware
//
#include <hip/hip_runtime.h>
#include <hip/hip_bf16.h>


typedef __attribute__((ext_vector_type(16))) _Float16 v16h;
typedef __attribute__((ext_vector_type(8)))  _Float16 v8h;
typedef __attribute__((ext_vector_type(8)))  float    v8f;
typedef __attribute__((ext_vector_type(8)))  int      v8i;

#define M_DIM 1024
#define N_DIM 512
#define DS 256
#define DH 32
#define DP 128
#define LN_EPS 1e-5f

static __device__ __forceinline__ v16h cat8(v8h lo, v8h hi) {
  return __builtin_shufflevector(lo, hi, 0,1,2,3,4,5,6,7,8,9,10,11,12,13,14,15);
}

__global__ void ln_proj_kernel(const float* __restrict__ x,
                               const int*   __restrict__ mask,
                               const float* __restrict__ gamma,
                               const float* __restrict__ beta,
                               const float* __restrict__ Wl,
                               const float* __restrict__ bl,
                               const float* __restrict__ Wr,
                               const float* __restrict__ br,
                               _Float16* __restrict__ Lb,
                               _Float16* __restrict__ Rb,
                               unsigned char* __restrict__ mfT) {
  __shared__ float sXn[8][DS];
  const int tid  = threadIdx.x;
  const int w    = tid >> 5;
  const int lane = tid & 31;
  const int nn   = blockIdx.x & (N_DIM - 1);
  const int mm   = (blockIdx.x >> 9) * 8 + w;

  const float* xrow = x + ((size_t)mm * N_DIM + nn) * DS;
  float4 v0 = *(const float4*)(xrow + lane * 8);
  float4 v1 = *(const float4*)(xrow + lane * 8 + 4);
  float s1 = v0.x + v0.y + v0.z + v0.w + v1.x + v1.y + v1.z + v1.w;
  float s2 = v0.x*v0.x + v0.y*v0.y + v0.z*v0.z + v0.w*v0.w
           + v1.x*v1.x + v1.y*v1.y + v1.z*v1.z + v1.w*v1.w;
  #pragma unroll
  for (int d = 16; d >= 1; d >>= 1) {
    s1 += __shfl_xor(s1, d, 32);
    s2 += __shfl_xor(s2, d, 32);
  }
  const float mean = s1 * (1.0f / DS);
  const float var  = s2 * (1.0f / DS) - mean * mean;
  const float rs   = rsqrtf(var + LN_EPS);

  float4 g0 = *(const float4*)(gamma + lane * 8);
  float4 g1 = *(const float4*)(gamma + lane * 8 + 4);
  float4 b0 = *(const float4*)(beta  + lane * 8);
  float4 b1 = *(const float4*)(beta  + lane * 8 + 4);
  float* sx = &sXn[w][lane * 8];
  sx[0] = (v0.x - mean) * rs * g0.x + b0.x;
  sx[1] = (v0.y - mean) * rs * g0.y + b0.y;
  sx[2] = (v0.z - mean) * rs * g0.z + b0.z;
  sx[3] = (v0.w - mean) * rs * g0.w + b0.w;
  sx[4] = (v1.x - mean) * rs * g1.x + b1.x;
  sx[5] = (v1.y - mean) * rs * g1.y + b1.y;
  sx[6] = (v1.z - mean) * rs * g1.z + b1.z;
  sx[7] = (v1.w - mean) * rs * g1.w + b1.w;
  __syncthreads();

  float accL = bl[lane];
  float accR = br[lane];
  const float* xn = sXn[w];
  #pragma unroll 8
  for (int c = 0; c < DS; ++c) {
    const float xv = xn[c];
    accL = fmaf(xv, Wl[c * DH + lane], accL);
    accR = fmaf(xv, Wr[c * DH + lane], accR);
  }
  const float mf = (mask[(size_t)mm * N_DIM + nn] > 0) ? 1.0f : 0.0f;
  accL *= mf;
  accR *= mf;

  const size_t o = ((size_t)lane * N_DIM + nn) * M_DIM + mm;
  Lb[o] = (_Float16)accL;
  Rb[o] = (_Float16)accR;
  if (lane == 0) mfT[(size_t)nn * M_DIM + mm] = (unsigned char)(mf > 0.0f);
}

__global__ void cnt_kernel(const unsigned char* __restrict__ mfT,
                           float* __restrict__ dinv) {
  const int lane = threadIdx.x & 31;
  const int lh = lane & 15, hb = lane >> 4;
  const int i0 = (blockIdx.x >> 5) * 16;
  const int j0 = (blockIdx.x & 31) * 16;

  v8i acc = {0,0,0,0,0,0,0,0};
  for (int k0 = 0; k0 < M_DIM; k0 += 64) {
    const unsigned char* ar = mfT + (size_t)(i0 + lh) * M_DIM + k0 + hb * 16;
    uint2 a0 = *(const uint2*)(ar);
    uint2 a1 = *(const uint2*)(ar + 8);
    uint2 a2 = *(const uint2*)(ar + 32);
    uint2 a3 = *(const uint2*)(ar + 40);
    v8i A; A[0]=(int)a0.x; A[1]=(int)a0.y; A[2]=(int)a1.x; A[3]=(int)a1.y;
           A[4]=(int)a2.x; A[5]=(int)a2.y; A[6]=(int)a3.x; A[7]=(int)a3.y;
    const unsigned char* br = mfT + (size_t)(j0 + lh) * M_DIM + k0 + hb * 16;
    uint4 q0 = *(const uint4*)(br);
    uint4 q1 = *(const uint4*)(br + 32);
    v8i B; B[0]=(int)q0.x; B[1]=(int)q0.y; B[2]=(int)q0.z; B[3]=(int)q0.w;
           B[4]=(int)q1.x; B[5]=(int)q1.y; B[6]=(int)q1.z; B[7]=(int)q1.w;
    acc = __builtin_amdgcn_wmma_i32_16x16x64_iu8(false, A, false, B, acc,
                                                 false, false);
  }
  #pragma unroll
  for (int r = 0; r < 8; ++r) {
    const float cnt = (float)acc[r];
    dinv[(size_t)(i0 + r + hb * 8) * N_DIM + j0 + lh] =
        1.0f / fmaxf(cnt + LN_EPS, 1.0f);
  }
}

#define LDSP 40

__global__ void outer_proj_kernel(const _Float16* __restrict__ Lb,
                                  const _Float16* __restrict__ Rb,
                                  const float* __restrict__ dinv,
                                  const _Float16* __restrict__ WoT,
                                  const float* __restrict__ bo,
                                  float* __restrict__ out) {
  __shared__ _Float16 sTile[1024 * LDSP];
  const int tid = threadIdx.x;
  const int w = tid >> 5, lane = tid & 31;
  const int lh = lane & 15, hb = lane >> 4;
  const int i0 = (blockIdx.x >> 4) * 32;
  const int j0 = (blockIdx.x & 15) * 32;

  v8f acc[4][2][2];
  #pragma unroll
  for (int q = 0; q < 4; ++q)
    #pragma unroll
    for (int ti = 0; ti < 2; ++ti)
      #pragma unroll
      for (int tj = 0; tj < 2; ++tj)
        #pragma unroll
        for (int r = 0; r < 8; ++r) acc[q][ti][tj][r] = 0.0f;

  const int aoff = hb ? 8 : 0;
  const int boff = hb ? 8 : 0;

  for (int k0 = 0; k0 < M_DIM; k0 += 32) {
    #pragma unroll
    for (int q = 0; q < 4; ++q) {
      const int h = w * 4 + q;
      const _Float16* ap =
          Lb + ((size_t)(h * N_DIM + i0 + lh)) * M_DIM + k0 + aoff;
      v16h A0 = cat8(*(const v8h*)ap, *(const v8h*)(ap + 16));
      v16h A1 = cat8(*(const v8h*)(ap + 16 * M_DIM),
                     *(const v8h*)(ap + 16 * M_DIM + 16));
      const _Float16* bp =
          Rb + ((size_t)(h * N_DIM + j0 + lh)) * M_DIM + k0 + boff;
      v16h B0 = cat8(*(const v8h*)bp, *(const v8h*)(bp + 16));
      v16h B1 = cat8(*(const v8h*)(bp + 16 * M_DIM),
                     *(const v8h*)(bp + 16 * M_DIM + 16));
      if (k0 + 32 < M_DIM) {
        __builtin_prefetch(ap + 32, 0, 1);
        __builtin_prefetch(bp + 32, 0, 1);
      }
      acc[q][0][0] = __builtin_amdgcn_wmma_f32_16x16x32_f16(
          false, A0, false, B0, (short)0, acc[q][0][0], false, false);
      acc[q][0][1] = __builtin_amdgcn_wmma_f32_16x16x32_f16(
          false, A0, false, B1, (short)0, acc[q][0][1], false, false);
      acc[q][1][0] = __builtin_amdgcn_wmma_f32_16x16x32_f16(
          false, A1, false, B0, (short)0, acc[q][1][0], false, false);
      acc[q][1][1] = __builtin_amdgcn_wmma_f32_16x16x32_f16(
          false, A1, false, B1, (short)0, acc[q][1][1], false, false);
    }
  }

  #pragma unroll
  for (int ti = 0; ti < 2; ++ti)
    #pragma unroll
    for (int tj = 0; tj < 2; ++tj)
      #pragma unroll
      for (int r = 0; r < 8; ++r) {
        const int il = ti * 16 + r + hb * 8;
        const int jl = tj * 16 + lh;
        const float dv = dinv[(size_t)(i0 + il) * N_DIM + j0 + jl];
        const int p = il * 32 + jl;
        #pragma unroll
        for (int q = 0; q < 4; ++q)
          sTile[p * LDSP + w * 4 + q] = (_Float16)(acc[q][ti][tj][r] * dv);
      }
  __syncthreads();

  #pragma unroll
  for (int s = 0; s < 8; ++s) {
    const int mt = w * 8 + s;
    const _Float16* sa = &sTile[(mt * 16 + lh) * LDSP + aoff];
    v16h a = cat8(*(const v8h*)sa, *(const v8h*)(sa + 16));
    #pragma unroll
    for (int nt = 0; nt < 8; ++nt) {
      const _Float16* wr = WoT + (nt * 16 + lh) * DH + boff;
      v16h b = cat8(*(const v8h*)wr, *(const v8h*)(wr + 16));
      const float bias = bo[nt * 16 + lh];
      v8f c;
      #pragma unroll
      for (int r = 0; r < 8; ++r) c[r] = bias;
      v8f d = __builtin_amdgcn_wmma_f32_16x16x32_f16(
          false, a, false, b, (short)0, c, false, false);
      #pragma unroll
      for (int r = 0; r < 8; ++r) {
        const int p = mt * 16 + r + hb * 8;
        out[((size_t)(i0 + (p >> 5)) * N_DIM + j0 + (p & 31)) * DP
            + nt * 16 + lh] = d[r];
      }
    }
  }
}

__global__ void prep_wot_kernel(const float* __restrict__ Wo,
                                _Float16* __restrict__ WoT) {
  const int t = blockIdx.x * 256 + threadIdx.x;
  if (t < DH * DP) {
    const int k = t >> 7, n = t & 127;
    WoT[n * DH + k] = (_Float16)Wo[k * DP + n];
  }
}

extern "C" void kernel_launch(void* const* d_in, const int* in_sizes, int n_in,
                              void* d_out, int out_size, void* d_ws, size_t ws_size,
                              hipStream_t stream) {
  const float* x     = (const float*)d_in[0];
  const int*   mask  = (const int*)  d_in[1];
  const float* gamma = (const float*)d_in[2];
  const float* beta  = (const float*)d_in[3];
  const float* Wl    = (const float*)d_in[4];
  const float* bl    = (const float*)d_in[5];
  const float* Wr    = (const float*)d_in[6];
  const float* br    = (const float*)d_in[7];
  const float* Wo    = (const float*)d_in[8];
  const float* bo    = (const float*)d_in[9];
  float* out = (float*)d_out;

  char* ws = (char*)d_ws;
  const size_t szLR = (size_t)DH * N_DIM * M_DIM * sizeof(_Float16);
  _Float16* Lb        = (_Float16*)(ws);
  _Float16* Rb        = (_Float16*)(ws + szLR);
  unsigned char* mfT  = (unsigned char*)(ws + 2 * szLR);
  float* dinv         = (float*)(ws + 2 * szLR + (size_t)N_DIM * M_DIM);
  _Float16* WoT       = (_Float16*)(ws + 2 * szLR + (size_t)N_DIM * M_DIM
                                       + (size_t)N_DIM * N_DIM * sizeof(float));

  prep_wot_kernel<<<(DH * DP + 255) / 256, 256, 0, stream>>>(Wo, WoT);
  ln_proj_kernel<<<N_DIM * (M_DIM / 8), 256, 0, stream>>>(
      x, mask, gamma, beta, Wl, bl, Wr, br, Lb, Rb, mfT);
  cnt_kernel<<<(N_DIM / 16) * (N_DIM / 16), 32, 0, stream>>>(mfT, dinv);
  outer_proj_kernel<<<(N_DIM / 32) * (N_DIM / 32), 256, 0, stream>>>(
      Lb, Rb, dinv, WoT, bo, out);
}
